// SpatialTransformerModel_70806830842265
// MI455X (gfx1250) — matrix operand order repaired and checked by reading, not compiled, not run on hardware
//
#include <hip/hip_runtime.h>
#include <hip/hip_bf16.h>
#include <math.h>

// ---------------------------------------------------------------------------
// SpatialTransformer forward for MI455X (gfx1250), wave32 + WMMA f16.
//   Stage 1: H=[X|STE] (f32) -> Q,K,V = relu(H@W+b) stored f16 in workspace
//   Stage 2: per-(b,t,head) flash attention over N=512 (d=8 zero-padded to K=32)
//            Q/K staged into LDS with GLOBAL_LOAD_ASYNC_TO_LDS_B128 (ASYNCcnt)
//   Stage 3: out = relu(AO@W_o1+b1)@W_o2+b2 -> f32 d_out
// Workspace: 4 x (49152*64) f16 buffers = ~25.2 MB.
// ---------------------------------------------------------------------------

typedef __attribute__((ext_vector_type(16))) _Float16 v16h;
typedef __attribute__((ext_vector_type(8)))  float    v8f;

#define RTOT (8 * 12 * 512)   // 49152 rows of the flattened [B,T,N] axis

union FragH { v16h v; unsigned u[8]; };

__device__ __forceinline__ unsigned pk2(float a, float b) {
  union { _Float16 h[2]; unsigned u; } x;
  x.h[0] = (_Float16)a; x.h[1] = (_Float16)b; return x.u;
}

// A-fragment (16x32 f16): dword v of lane (hi = lane>>4) holds K = akoff, akoff+1
// per CDNA5 ISA 7.12.2 (VGPR0..3: K=0..7 (+8 for hi), VGPR4..7: K=16..23 (+8 for hi))
__device__ __forceinline__ int akoff(int v, int hi) {
  return ((v < 4) ? (2 * v) : (2 * v + 8)) + hi * 8;
}
// B-fragment (32x16 f16): same K order as the A-fragment (akoff)
__device__ __forceinline__ int bkoff(int v, int hi) {
  return ((v < 4) ? (2 * v) : (2 * v + 8)) + hi * 8;
}

// Async 16-byte memory -> LDS copy (CDNA5 VGLOBAL async op, tracked by ASYNCcnt)
__device__ __forceinline__ void async_copy_b128(unsigned lds_off, const void* gaddr) {
  asm volatile("global_load_async_to_lds_b128 %0, %1, off"
               :: "v"(lds_off), "v"(gaddr) : "memory");
}
__device__ __forceinline__ void wait_asynccnt0() {
  asm volatile("s_wait_asynccnt 0" ::: "memory");
}

// ---------------------------------------------------------------------------
// Stage 1: QKV projection.  H[r][k] = (k<64 ? X[r][k] : STE[r][k-64]), f32->f16.
// Weights staged transposed in LDS: WT[n][k] = W[k][n] (f16) so B-fragment
// dwords are contiguous.  Each wave computes one 16-row tile for q,k,v.
// ---------------------------------------------------------------------------
__global__ __launch_bounds__(256) void qkv_kernel(
    const float* __restrict__ X, const float* __restrict__ STE,
    const float* __restrict__ Wq, const float* __restrict__ bq,
    const float* __restrict__ Wk, const float* __restrict__ bk,
    const float* __restrict__ Wv, const float* __restrict__ bv,
    _Float16* __restrict__ Qh, _Float16* __restrict__ Kh, _Float16* __restrict__ Vh)
{
  __shared__ _Float16 WT[3][64][128];   // 48 KB
  const int tid = threadIdx.x;
  for (int i = tid; i < 3 * 64 * 128; i += 256) {
    int w = i >> 13;
    int r = i & 8191;
    int n = r >> 7, k = r & 127;
    const float* W = (w == 0) ? Wq : (w == 1) ? Wk : Wv;
    WT[w][n][k] = (_Float16)W[k * 64 + n];
  }
  __syncthreads();

  const int lane = tid & 31, wave = tid >> 5;
  const int m = lane & 15, hi = lane >> 4;
  const int row0 = (blockIdx.x * 8 + wave) * 16;
  const int rowA = row0 + m;

  v8f acc[3][4] = {};

  for (int ks = 0; ks < 4; ++ks) {
    FragH a;
    for (int v = 0; v < 8; ++v) {
      int k0 = akoff(v, hi) + 32 * ks;
      const float* src = (k0 < 64) ? (X + (size_t)rowA * 64 + k0)
                                   : (STE + (size_t)rowA * 64 + (k0 - 64));
      float2 f = *reinterpret_cast<const float2*>(src);
      a.u[v] = pk2(f.x, f.y);
    }
    for (int w = 0; w < 3; ++w) {
      for (int nt = 0; nt < 4; ++nt) {
        FragH b;
        int n = m + 16 * nt;
        for (int v = 0; v < 8; ++v) {
          int kb = bkoff(v, hi) + 32 * ks;
          b.u[v] = *reinterpret_cast<const unsigned*>(&WT[w][n][kb]);
        }
        acc[w][nt] = __builtin_amdgcn_wmma_f32_16x16x32_f16(
            false, a.v, false, b.v, (short)0, acc[w][nt], false, false);
      }
    }
  }
  // bias + relu + f16 store (D-layout: lane holds column n, VGPR j -> row j+8*hi)
  for (int w = 0; w < 3; ++w) {
    const float* bias = (w == 0) ? bq : (w == 1) ? bk : bv;
    _Float16* out = (w == 0) ? Qh : (w == 1) ? Kh : Vh;
    for (int nt = 0; nt < 4; ++nt) {
      int n = m + 16 * nt;
      float bn = bias[n];
      for (int j = 0; j < 8; ++j) {
        float x = acc[w][nt][j] + bn;
        x = x > 0.f ? x : 0.f;
        int r = row0 + j + 8 * hi;
        out[(size_t)r * 64 + n] = (_Float16)x;
      }
    }
  }
}

// ---------------------------------------------------------------------------
// Stage 2: flash attention, one workgroup per (b,t,head); 8 waves, each wave
// owns 4 query blocks of 16 rows; streams keys in blocks of 32.
// LDS: q[512][8], k[512][8], vT[8][512] f16 + per-wave P staging [16][32].
// Q/K rows (16B each) are copied memory->LDS with async B128 transfers.
// ---------------------------------------------------------------------------
__global__ __launch_bounds__(256) void attn_kernel(
    const _Float16* __restrict__ Qh, const _Float16* __restrict__ Kh,
    const _Float16* __restrict__ Vh, _Float16* __restrict__ AO)
{
  __shared__ __align__(16) _Float16 sQ[512][8];
  __shared__ __align__(16) _Float16 sK[512][8];
  __shared__ __align__(16) _Float16 sVT[8][512];
  __shared__ __align__(16) _Float16 sP[8][16 * 32];

  const int tid = threadIdx.x;
  const int head = blockIdx.x;
  const int bt = head >> 3, kh = head & 7;
  const size_t base = (size_t)bt * 512;

  for (int r = tid; r < 512; r += 256) {
    size_t g = (base + r) * 64 + kh * 8;
    // Q and K: straight async memory->LDS (one B128 per 8-half row)
    async_copy_b128((unsigned)(uintptr_t)&sQ[r][0], (const void*)(Qh + g));
    async_copy_b128((unsigned)(uintptr_t)&sK[r][0], (const void*)(Kh + g));
    // V must be transposed into sVT -> go through VGPRs
    _Float16 tmp[8];
    *reinterpret_cast<uint4*>(tmp) = *reinterpret_cast<const uint4*>(Vh + g);
    for (int dd = 0; dd < 8; ++dd) sVT[dd][r] = tmp[dd];
  }
  wait_asynccnt0();
  __syncthreads();

  const int lane = tid & 31, wave = tid >> 5;
  const int m = lane & 15, hi = lane >> 4;
  const float scale = 0.35355339059327373f;  // 1/sqrt(8)
  const v8f vz = {};
  _Float16* pw = &sP[wave][0];

  for (int qb = wave; qb < 32; qb += 8) {
    FragH aq;  // q tile as A-fragment, K=8 zero-padded to 32
    for (int v = 0; v < 8; ++v)
      aq.u[v] = (hi == 0 && v < 4)
                    ? *reinterpret_cast<const unsigned*>(&sQ[qb * 16 + m][2 * v]) : 0u;

    float mrun[8], lrun[8];
    for (int j = 0; j < 8; ++j) { mrun[j] = -INFINITY; lrun[j] = 0.f; }
    v8f accO = vz;

    for (int c = 0; c < 512; c += 32) {
      // two 16x16 score tiles: keys c..c+15 and c+16..c+31 (B lane = key)
      FragH bk0, bk1;
      for (int v = 0; v < 8; ++v) {
        bk0.u[v] = (hi == 0 && v < 4)
                       ? *reinterpret_cast<const unsigned*>(&sK[c + m][2 * v]) : 0u;
        bk1.u[v] = (hi == 0 && v < 4)
                       ? *reinterpret_cast<const unsigned*>(&sK[c + 16 + m][2 * v]) : 0u;
      }
      v8f s0 = __builtin_amdgcn_wmma_f32_16x16x32_f16(false, aq.v, false, bk0.v, (short)0, vz, false, false);
      v8f s1 = __builtin_amdgcn_wmma_f32_16x16x32_f16(false, aq.v, false, bk1.v, (short)0, vz, false, false);

      float p0[8], p1[8];
      for (int j = 0; j < 8; ++j) {
        float a0 = s0[j] * scale, a1 = s1[j] * scale;
        float t = fmaxf(a0, a1);
        for (int off = 1; off < 16; off <<= 1) t = fmaxf(t, __shfl_xor(t, off, 32));
        float mnew = fmaxf(mrun[j], t);
        p0[j] = __expf(a0 - mnew);
        p1[j] = __expf(a1 - mnew);
        float rs = p0[j] + p1[j];
        for (int off = 1; off < 16; off <<= 1) rs += __shfl_xor(rs, off, 32);
        float alpha = __expf(mrun[j] - mnew);
        lrun[j] = lrun[j] * alpha + rs;
        mrun[j] = mnew;
        accO[j] *= alpha;
      }
      // P tile D-layout -> LDS -> reload as A-fragment (wave-private region)
      for (int j = 0; j < 8; ++j) {
        int rr = j + 8 * hi;
        pw[rr * 32 + m]      = (_Float16)p0[j];
        pw[rr * 32 + 16 + m] = (_Float16)p1[j];
      }
      asm volatile("s_wait_dscnt 0" ::: "memory");
      FragH ap, bvv;
      for (int v = 0; v < 8; ++v)
        ap.u[v] = *reinterpret_cast<const unsigned*>(&pw[m * 32 + akoff(v, hi)]);
      for (int v = 0; v < 8; ++v)
        bvv.u[v] = (m < 8)
                       ? *reinterpret_cast<const unsigned*>(&sVT[m][c + bkoff(v, hi)]) : 0u;
      accO = __builtin_amdgcn_wmma_f32_16x16x32_f16(false, ap.v, false, bvv.v, (short)0, accO, false, false);
    }
    // normalize and store: lane column m = feature dd (<8 valid)
    if (m < 8) {
      for (int j = 0; j < 8; ++j) {
        int r = qb * 16 + j + 8 * hi;
        AO[(base + r) * 64 + kh * 8 + m] = (_Float16)(accO[j] / lrun[j]);
      }
    }
  }
}

// ---------------------------------------------------------------------------
// Stage 3: fused FFN: Y = relu(AO@W1+b1)@W2+b2, f32 output.
// ---------------------------------------------------------------------------
__global__ __launch_bounds__(256) void ffn_kernel(
    const _Float16* __restrict__ AO,
    const float* __restrict__ W1, const float* __restrict__ b1,
    const float* __restrict__ W2, const float* __restrict__ b2,
    float* __restrict__ OUT)
{
  __shared__ _Float16 WT1[64][64];
  __shared__ _Float16 WT2[64][64];
  __shared__ __align__(16) _Float16 sY[8][16 * 64];

  const int tid = threadIdx.x;
  for (int i = tid; i < 4096; i += 256) {
    int n = i >> 6, k = i & 63;
    WT1[n][k] = (_Float16)W1[k * 64 + n];
    WT2[n][k] = (_Float16)W2[k * 64 + n];
  }
  __syncthreads();

  const int lane = tid & 31, wave = tid >> 5;
  const int m = lane & 15, hi = lane >> 4;
  const int row0 = (blockIdx.x * 8 + wave) * 16;
  _Float16* yw = &sY[wave][0];

  v8f acc1[4] = {};
  for (int ks = 0; ks < 2; ++ks) {
    FragH a;
    for (int v = 0; v < 8; ++v) {
      int kb = akoff(v, hi) + 32 * ks;
      a.u[v] = *reinterpret_cast<const unsigned*>(AO + (size_t)(row0 + m) * 64 + kb);
    }
    for (int nt = 0; nt < 4; ++nt) {
      FragH b;
      int n = m + 16 * nt;
      for (int v = 0; v < 8; ++v)
        b.u[v] = *reinterpret_cast<const unsigned*>(&WT1[n][bkoff(v, hi) + 32 * ks]);
      acc1[nt] = __builtin_amdgcn_wmma_f32_16x16x32_f16(
          false, a.v, false, b.v, (short)0, acc1[nt], false, false);
    }
  }
  for (int nt = 0; nt < 4; ++nt) {
    int n = m + 16 * nt;
    float bn = b1[n];
    for (int j = 0; j < 8; ++j) {
      float x = acc1[nt][j] + bn;
      x = x > 0.f ? x : 0.f;
      yw[(j + 8 * hi) * 64 + n] = (_Float16)x;
    }
  }
  asm volatile("s_wait_dscnt 0" ::: "memory");

  v8f acc2[4] = {};
  for (int ks = 0; ks < 2; ++ks) {
    FragH a;
    for (int v = 0; v < 8; ++v)
      a.u[v] = *reinterpret_cast<const unsigned*>(&yw[m * 64 + akoff(v, hi) + 32 * ks]);
    for (int nt = 0; nt < 4; ++nt) {
      FragH b;
      int n = m + 16 * nt;
      for (int v = 0; v < 8; ++v)
        b.u[v] = *reinterpret_cast<const unsigned*>(&WT2[n][bkoff(v, hi) + 32 * ks]);
      acc2[nt] = __builtin_amdgcn_wmma_f32_16x16x32_f16(
          false, a.v, false, b.v, (short)0, acc2[nt], false, false);
    }
  }
  for (int nt = 0; nt < 4; ++nt) {
    int n = m + 16 * nt;
    float bn = b2[n];
    for (int j = 0; j < 8; ++j)
      OUT[(size_t)(row0 + j + 8 * hi) * 64 + n] = acc2[nt][j] + bn;
  }
}

// ---------------------------------------------------------------------------
extern "C" void kernel_launch(void* const* d_in, const int* in_sizes, int n_in,
                              void* d_out, int out_size, void* d_ws, size_t ws_size,
                              hipStream_t stream) {
  const float* X   = (const float*)d_in[0];
  const float* STE = (const float*)d_in[1];
  const float* Wq  = (const float*)d_in[2];
  const float* bq  = (const float*)d_in[3];
  const float* Wk  = (const float*)d_in[4];
  const float* bk  = (const float*)d_in[5];
  const float* Wv  = (const float*)d_in[6];
  const float* bv  = (const float*)d_in[7];
  const float* W1  = (const float*)d_in[8];
  const float* b1  = (const float*)d_in[9];
  const float* W2  = (const float*)d_in[10];
  const float* b2  = (const float*)d_in[11];
  (void)in_sizes; (void)n_in; (void)out_size; (void)ws_size;

  const size_t RD = (size_t)RTOT * 64;
  _Float16* Qh = (_Float16*)d_ws;
  _Float16* Kh = Qh + RD;
  _Float16* Vh = Kh + RD;
  _Float16* AO = Vh + RD;

  qkv_kernel<<<RTOT / 128, 256, 0, stream>>>(X, STE, Wq, bq, Wk, bk, Wv, bv, Qh, Kh, Vh);
  attn_kernel<<<8 * 12 * 8, 256, 0, stream>>>(Qh, Kh, Vh, AO);
  ffn_kernel<<<RTOT / 128, 256, 0, stream>>>(AO, W1, b1, W2, b2, (float*)d_out);
}
